// GNNRecommender_63050119905949
// MI455X (gfx1250) — matrix operand order repaired and checked by reading, not compiled, not run on hardware
//
#include <hip/hip_runtime.h>
#include <hip/hip_bf16.h>
#include <math.h>

typedef __attribute__((ext_vector_type(16))) _Float16 v16h;
typedef __attribute__((ext_vector_type(8)))  float    v8f;

#define EMB 64
#define HID 128
#define NUSERS 100000
#define LN_EPS 1e-5f

// ----------------- degree / symmetric norm -----------------
__global__ void k_deg_init(float* deg, int n) {
  int i = blockIdx.x * blockDim.x + threadIdx.x;
  if (i < n) deg[i] = 1.0f;                       // self-loop
}
__global__ void k_deg_scatter(const int* __restrict__ dst, float* deg, int e) {
  int i = blockIdx.x * blockDim.x + threadIdx.x;
  if (i < e) atomicAdd(&deg[dst[i]], 1.0f);
}
__global__ void k_rsqrt_inplace(float* d, int n) {
  int i = blockIdx.x * blockDim.x + threadIdx.x;
  if (i < n) d[i] = rsqrtf(d[i]);                 // deg -> dinv
}

// ----------------- weight pre-pack: f32 [K,Nc] -> f16 B-fragments ---------
// Fragment (tn, kb) stored as 32 lanes x 16 contiguous halves:
//   P[((tn*nkb + kb)*32 + lane)*16 + j] = W[(kb*32 + 8*(lane>=16) + 16*(j>=8) + (j&7))*Nc + tn*16 + (lane&15)]
__global__ void k_pack_bfrag(const float* __restrict__ W, _Float16* __restrict__ P,
                             int K, int Nc) {
  int i = blockIdx.x * blockDim.x + threadIdx.x;
  if (i >= K * Nc) return;
  int j    = i & 15;
  int lane = (i >> 4) & 31;
  int frag = i >> 9;
  int nkb  = K >> 5;
  int kb   = frag % nkb;
  int tn   = frag / nkb;
  int col  = (tn << 4) + (lane & 15);
  int krow = (kb << 5) + ((lane >> 4) << 3) + ((j >> 3) << 4) + (j & 7);
  P[i] = (_Float16)W[(size_t)krow * Nc + col];
}

// ----------------- WMMA f16->f32 GEMM: C[M,Nc] = (relu?)A[M,K] @ B[K,Nc] ---
// One wave per 16 x (TN*16) strip of C.  A row-major f32, loaded as 4x b128
// per k-step and converted; B pre-packed f16 fragments (2x b128 per tile).
//   A 16x32 f16 : row = lane&15,  K = 8*(lane>=16) + 16*(j>=8) + (j&7)
//   C 16x16 f32 : row = r + 8*(lane>=16), col = lane&15
template<int TN, bool RELU>
__global__ void k_wmma_gemm(const float* __restrict__ A, const _Float16* __restrict__ P,
                            float* __restrict__ C, int M, int K, int Nc) {
  const int wave   = (blockIdx.x * blockDim.x + threadIdx.x) >> 5;
  const int lane   = threadIdx.x & 31;
  const int nstrip = Nc / (TN * 16);
  const int tm     = wave / nstrip;
  const int ts     = wave - tm * nstrip;
  if (tm >= (M >> 4)) return;                     // wave-uniform: EXEC stays full

  const int    rowA = (tm << 4) + (lane & 15);
  const int    khiA = (lane >> 4) << 3;           // 0 or 8
  const float* Ar   = A + (size_t)rowA * K + khiA;
  const int    nkb  = K >> 5;
  const _Float16* Pb = P + (size_t)(ts * TN) * nkb * 512 + lane * 16;

  v8f acc[TN] = {};
  for (int kb = 0; kb < nkb; ++kb) {
    if (kb + 1 < nkb) __builtin_prefetch(Ar + (kb + 1) * 32, 0, 0);
    const float4 f0 = *(const float4*)(Ar + kb * 32);
    const float4 f1 = *(const float4*)(Ar + kb * 32 + 4);
    const float4 f2 = *(const float4*)(Ar + kb * 32 + 16);
    const float4 f3 = *(const float4*)(Ar + kb * 32 + 20);
    float af[16] = { f0.x, f0.y, f0.z, f0.w, f1.x, f1.y, f1.z, f1.w,
                     f2.x, f2.y, f2.z, f2.w, f3.x, f3.y, f3.z, f3.w };
    v16h a;
#pragma unroll
    for (int j = 0; j < 16; ++j) {
      float av = af[j];
      if (RELU) av = fmaxf(av, 0.0f);
      a[j] = (_Float16)av;
    }
#pragma unroll
    for (int t = 0; t < TN; ++t) {
      v16h b = *(const v16h*)(Pb + ((size_t)t * nkb + kb) * 512);
      acc[t] = __builtin_amdgcn_wmma_f32_16x16x32_f16(false, a, false, b,
                                                      (short)0, acc[t], false, false);
    }
  }
  const int rowC = (tm << 4) + ((lane >> 4) << 3);
#pragma unroll
  for (int t = 0; t < TN; ++t) {
    const int colC = (ts * TN + t) * 16 + (lane & 15);
#pragma unroll
    for (int r = 0; r < 8; ++r)
      C[(size_t)(rowC + r) * Nc + colC] = acc[t][r];
  }
}

// ----------------- GCN self-loop + bias:  out = xw * dinv^2 + b -----------
__global__ void k_selfloop_bias(const float* __restrict__ xw, const float* __restrict__ dinv,
                                const float* __restrict__ bias, float* __restrict__ out,
                                int rows, int cols) {
  int i = blockIdx.x * blockDim.x + threadIdx.x;
  if (i >= rows * cols) return;
  int r = i / cols, c = i - r * cols;
  float dv = dinv[r];
  out[i] = xw[i] * dv * dv + bias[c];
}

// ----------------- edge aggregation: one wave per edge --------------------
// out[dst] += xw[src] * dinv[src]*dinv[dst]   (coalesced row read, f32 atomics)
__global__ void k_edge_aggregate(const float* __restrict__ xw, const float* __restrict__ dinv,
                                 const int* __restrict__ src, const int* __restrict__ dst,
                                 float* out, int e, int cols) {
  int wave = (blockIdx.x * blockDim.x + threadIdx.x) >> 5;
  int lane = threadIdx.x & 31;
  if (wave >= e) return;
  int s = src[wave], d = dst[wave];
  float nrm = dinv[s] * dinv[d];
  const float* xs = xw + (size_t)s * cols;
  float* od = out + (size_t)d * cols;
  for (int c = lane; c < cols; c += 32)
    atomicAdd(&od[c], xs[c] * nrm);
}

// ----------------- gather user/item rows + concat -------------------------
__global__ void k_gather_concat(const float* __restrict__ h, const int* __restrict__ uidx,
                                const int* __restrict__ iidx, float* __restrict__ cf, int b) {
  int i = blockIdx.x * blockDim.x + threadIdx.x;
  if (i >= b * 2 * EMB) return;
  int row = i >> 7, c = i & 127;
  int node = (c < EMB) ? (uidx[row] - 1) : (NUSERS + iidx[row] - 1);
  int cc   = (c < EMB) ? c : (c - EMB);
  cf[i] = h[(size_t)node * EMB + cc];
}

// ----------------- bias + LayerNorm + ReLU (in place), one wave per row ---
template<int COLS>
__global__ void k_bias_ln_relu(float* __restrict__ z, const float* __restrict__ bias,
                               const float* __restrict__ g, const float* __restrict__ be,
                               int rows) {
  int wave = (blockIdx.x * blockDim.x + threadIdx.x) >> 5;
  int lane = threadIdx.x & 31;
  if (wave >= rows) return;
  constexpr int PER = COLS / 32;
  float* zr = z + (size_t)wave * COLS;
  float v[PER]; float s = 0.f;
#pragma unroll
  for (int i = 0; i < PER; ++i) { int c = lane + i * 32; v[i] = zr[c] + bias[c]; s += v[i]; }
#pragma unroll
  for (int m = 16; m >= 1; m >>= 1) s += __shfl_xor(s, m, 32);
  float mu = s * (1.0f / COLS);
  float vs = 0.f;
#pragma unroll
  for (int i = 0; i < PER; ++i) { float d = v[i] - mu; vs += d * d; }
#pragma unroll
  for (int m = 16; m >= 1; m >>= 1) vs += __shfl_xor(vs, m, 32);
  float rstd = rsqrtf(vs * (1.0f / COLS) + LN_EPS);
#pragma unroll
  for (int i = 0; i < PER; ++i) {
    int c = lane + i * 32;
    zr[c] = fmaxf((v[i] - mu) * rstd * g[c] + be[c], 0.0f);
  }
}

// ----------------- head: sigmoid(z @ fcW3 + b), one wave per row ----------
__global__ void k_head(const float* __restrict__ z, const float* __restrict__ w,
                       const float* __restrict__ b, float* __restrict__ out, int rows) {
  int wave = (blockIdx.x * blockDim.x + threadIdx.x) >> 5;
  int lane = threadIdx.x & 31;
  if (wave >= rows) return;
  const float* zr = z + (size_t)wave * EMB;
  float s = zr[lane] * w[lane] + zr[lane + 32] * w[lane + 32];
#pragma unroll
  for (int m = 16; m >= 1; m >>= 1) s += __shfl_xor(s, m, 32);
  if (lane == 0) out[wave] = 1.0f / (1.0f + __expf(-(s + b[0])));
}

extern "C" void kernel_launch(void* const* d_in, const int* in_sizes, int n_in,
                              void* d_out, int out_size, void* d_ws, size_t ws_size,
                              hipStream_t stream) {
  const float* x    = (const float*)d_in[0];
  const int*   src  = (const int*)  d_in[1];
  const int*   dst  = (const int*)  d_in[2];
  const int*   uidx = (const int*)  d_in[3];
  const int*   iidx = (const int*)  d_in[4];
  const float* W1   = (const float*)d_in[5];
  const float* b1   = (const float*)d_in[6];
  const float* W2   = (const float*)d_in[7];
  const float* b2   = (const float*)d_in[8];
  const float* fcW1 = (const float*)d_in[9];
  const float* fcb1 = (const float*)d_in[10];
  const float* g1   = (const float*)d_in[11];
  const float* be1  = (const float*)d_in[12];
  const float* fcW2 = (const float*)d_in[13];
  const float* fcb2 = (const float*)d_in[14];
  const float* g2   = (const float*)d_in[15];
  const float* be2  = (const float*)d_in[16];
  const float* fcW3 = (const float*)d_in[17];
  const float* fcb3 = (const float*)d_in[18];

  const int Nn = in_sizes[0] / EMB;   // 150000 (divisible by 16)
  const int E  = in_sizes[1];         // 2,400,000
  const int B  = in_sizes[3];         // 16384

  // workspace layout (floats / halves), aggressively reused; all 32B aligned
  float* ws   = (float*)d_ws;
  float* dinv = ws;                            // [Nn]
  float* bufA = dinv + Nn;                     // [Nn*HID]  xw1 -> xw2
  float* bufB = bufA + (size_t)Nn * HID;       // [Nn*HID]  out1 -> out2
  float* bufC = bufB + (size_t)Nn * HID;       // [B*128]   concat -> z2raw/z2
  float* bufD = bufC + (size_t)B * 2 * EMB;    // [B*128]   z1raw/z1
  _Float16* pW1 = (_Float16*)(bufD + (size_t)B * HID);  // [64*128]
  _Float16* pW2 = pW1 + EMB * HID;                      // [128*64]
  _Float16* pF1 = pW2 + HID * EMB;                      // [128*128]
  _Float16* pF2 = pF1 + HID * HID;                      // [128*64]

  const int T = 256;
  auto nb = [](long long t, int bs) { return (int)((t + bs - 1) / bs); };

  // pack weights into f16 B-fragment layout (tiny, L2-resident thereafter)
  k_pack_bfrag<<<nb(EMB * HID, T), T, 0, stream>>>(W1,   pW1, EMB, HID);
  k_pack_bfrag<<<nb(HID * EMB, T), T, 0, stream>>>(W2,   pW2, HID, EMB);
  k_pack_bfrag<<<nb(HID * HID, T), T, 0, stream>>>(fcW1, pF1, HID, HID);
  k_pack_bfrag<<<nb(HID * EMB, T), T, 0, stream>>>(fcW2, pF2, HID, EMB);

  // degrees -> dinv
  k_deg_init     <<<nb(Nn, T), T, 0, stream>>>(dinv, Nn);
  k_deg_scatter  <<<nb(E,  T), T, 0, stream>>>(dst, dinv, E);
  k_rsqrt_inplace<<<nb(Nn, T), T, 0, stream>>>(dinv, Nn);

  // GCN layer 1: xw1 = x@W1 ; out1 = agg(xw1) + xw1*dinv^2 + b1
  long long w1 = (long long)(Nn / 16) * (HID / 64);
  k_wmma_gemm<4, false><<<nb(w1 * 32, T), T, 0, stream>>>(x, pW1, bufA, Nn, EMB, HID);
  k_selfloop_bias<<<nb((long long)Nn * HID, T), T, 0, stream>>>(bufA, dinv, b1, bufB, Nn, HID);
  k_edge_aggregate<<<nb((long long)E * 32, T), T, 0, stream>>>(bufA, dinv, src, dst, bufB, E, HID);

  // GCN layer 2: xw2 = relu(out1)@W2 ; out2 = agg(xw2) + xw2*dinv^2 + b2
  long long w2 = (long long)(Nn / 16) * (EMB / 64);
  k_wmma_gemm<4, true><<<nb(w2 * 32, T), T, 0, stream>>>(bufB, pW2, bufA, Nn, HID, EMB);
  k_selfloop_bias<<<nb((long long)Nn * EMB, T), T, 0, stream>>>(bufA, dinv, b2, bufB, Nn, EMB);
  k_edge_aggregate<<<nb((long long)E * 32, T), T, 0, stream>>>(bufA, dinv, src, dst, bufB, E, EMB);

  // gather + concat -> [B, 128]
  k_gather_concat<<<nb((long long)B * 2 * EMB, T), T, 0, stream>>>(bufB, uidx, iidx, bufC, B);

  // MLP head
  long long f1 = (long long)(B / 16) * (HID / 64);
  k_wmma_gemm<4, false><<<nb(f1 * 32, T), T, 0, stream>>>(bufC, pF1, bufD, B, 2 * EMB, HID);
  k_bias_ln_relu<HID><<<nb((long long)B * 32, T), T, 0, stream>>>(bufD, fcb1, g1, be1, B);

  long long f2 = (long long)(B / 16) * (EMB / 64);
  k_wmma_gemm<4, false><<<nb(f2 * 32, T), T, 0, stream>>>(bufD, pF2, bufC, B, HID, EMB);
  k_bias_ln_relu<EMB><<<nb((long long)B * 32, T), T, 0, stream>>>(bufC, fcb2, g2, be2, B);

  k_head<<<nb((long long)B * 32, T), T, 0, stream>>>(bufC, fcW3, fcb3, (float*)d_out, B);
}
